// CoarseMatching_18416819765264
// MI455X (gfx1250) — matrix operand order repaired and checked by reading, not compiled, not run on hardware
//
#include <hip/hip_runtime.h>

// ---------------------------------------------------------------------------
// CoarseMatching flow head for MI455X (gfx1250), wave32 + WMMA bf16 +
// async global->LDS staging (GLOBAL_LOAD_ASYNC_TO_LDS_B128 / ASYNCcnt).
//   1) proj_kernel:  g = (feat @ W^T + b) * scale  -> bf16 workspace
//        f0 gets scale = (1/C)*log2e (folds corr 1/sqrt(C) + softmax log2e),
//        f1 gets scale = 1/sqrt(C)   => WMMA acc == corr * log2e.
//   2) corr_flow_kernel: f1 s-tiles staged to double-buffered LDS by the
//        async engine, WMMA from LDS, log2-domain online softmax,
//        expected coords -> flow -> border mask.
// ---------------------------------------------------------------------------

#define L_TOT 9216   // 96*96
#define C_DIM 256
#define HW    96
#define NB    2
#define STILES (L_TOT / 16)          // 576
#define LOG2E 1.4426950408889634f
#define INV_SQRT_C 0.0625f           // 1/sqrt(256)

// LDS tile: 16 rows x 512B, rows padded to 528B (132 dw = 4 mod 64 banks)
#define ROW_BYTES   512
#define ROW_PAD_B   528
#define ROW_PAD_E   264              // __bf16 elements per padded row
#define TILE_BYTES  8192             // 16 * 512, contiguous in f1

typedef __attribute__((ext_vector_type(16))) __bf16 v16bf;
typedef __attribute__((ext_vector_type(8)))  __bf16 v8bf;
typedef __attribute__((ext_vector_type(8)))  float  v8f;

static __device__ __forceinline__ __bf16 f2bf(float f) {
  return static_cast<__bf16>(f);     // v_cvt_pk_bf16_f32 pairs
}

// Flat/generic LDS pointers carry the LDS byte offset in their low 32 bits;
// the async-DMA VDST operand wants exactly that offset (HW adds LDS_BASE).
static __device__ __forceinline__ unsigned lds_off_u32(const void* p) {
  return (unsigned)(unsigned long long)(uintptr_t)p;
}

// Async copy 16B global -> LDS, tracked by ASYNCcnt (GV addressing mode).
static __device__ __forceinline__ void async_copy_b128(unsigned lds, const void* g) {
  asm volatile("global_load_async_to_lds_b128 %0, %1, off"
               :: "v"(lds), "v"(g)
               : "memory");
}

// ---------------------------------------------------------------------------
// Kernel 1: projection GEMM  out[m,n] = (sum_c feat[m,c]*W[n,c] + b[n])*scale
// ---------------------------------------------------------------------------
__global__ __launch_bounds__(256) void proj_kernel(
    const float* __restrict__ feat, const float* __restrict__ W,
    const float* __restrict__ bias, __bf16* __restrict__ outh, float scale) {
  const int lane = threadIdx.x & 31;
  const int wave = threadIdx.x >> 5;
  const int half = lane >> 4;
  const int l15  = lane & 15;
  const int row_base = blockIdx.x * 128 + wave * 16;
  const int col_base = blockIdx.y * 16;

  const float* aptr = feat + (size_t)(row_base + l15) * C_DIM;
  const float* bptr = W    + (size_t)(col_base + l15) * C_DIM;

  v8f acc = {};
#pragma unroll
  for (int kk = 0; kk < 8; ++kk) {
    const int k0 = kk * 32;
    float ta[16] __attribute__((aligned(16)));
    *(float4*)(ta + 0)  = *(const float4*)(aptr + k0 + half * 16);
    *(float4*)(ta + 4)  = *(const float4*)(aptr + k0 + half * 16 + 4);
    *(float4*)(ta + 8)  = *(const float4*)(aptr + k0 + half * 16 + 8);
    *(float4*)(ta + 12) = *(const float4*)(aptr + k0 + half * 16 + 12);
    float tb[16] __attribute__((aligned(16)));
    *(float4*)(tb + 0)  = *(const float4*)(bptr + k0 + half * 16);
    *(float4*)(tb + 4)  = *(const float4*)(bptr + k0 + half * 16 + 4);
    *(float4*)(tb + 8)  = *(const float4*)(bptr + k0 + half * 16 + 8);
    *(float4*)(tb + 12) = *(const float4*)(bptr + k0 + half * 16 + 12);
    v16bf a, bf;
#pragma unroll
    for (int j = 0; j < 16; ++j) { a[j] = f2bf(ta[j]); bf[j] = f2bf(tb[j]); }
    acc = __builtin_amdgcn_wmma_f32_16x16x32_bf16(
        false, a, false, bf, (short)0, acc, false, false);
  }
  const float bv = bias[col_base + l15];
#pragma unroll
  for (int v = 0; v < 8; ++v) {
    const int orow = row_base + v + 8 * half;
    outh[(size_t)orow * C_DIM + (col_base + l15)] = f2bf((acc[v] + bv) * scale);
  }
}

// ---------------------------------------------------------------------------
// Kernel 2: fused correlation + online softmax + expected coords -> flow.
// Block = 128 threads (4 waves) = 64 queries; each wave owns 16 queries
// (resident B fragments).  All waves sweep all 576 f1 s-tiles; each 8KB tile
// is async-DMA'd once into double-buffered LDS and consumed by all 4 waves
// via ds_load_b128.  Scores are pre-scaled by log2e; each lane's 8 scores are
// consecutive s starting at an 8-aligned s0 and 96%8==0, so y is constant
// per tile and x = rem + v is linear.
// ---------------------------------------------------------------------------
__global__ __launch_bounds__(128) void corr_flow_kernel(
    const __bf16* __restrict__ f0h, const __bf16* __restrict__ f1h,
    float* __restrict__ out) {
  const int tid  = threadIdx.x;
  const int lane = tid & 31;
  const int wave = tid >> 5;
  const int half = lane >> 4;
  const int q    = lane & 15;
  const int b    = blockIdx.y;
  const int qblk = blockIdx.x;            // 64-query strip

  const __bf16* f0base = f0h + ((size_t)b * L_TOT + (size_t)qblk * 64) * C_DIM;
  const char*   f1tile = (const char*)(f1h + (size_t)b * L_TOT * C_DIM);

  __shared__ __attribute__((aligned(16))) __bf16 tile[2][16 * ROW_PAD_E];

  // Resident B fragments: query (wave*16+q) as column, K = kk*32+half*16+0..15
  v16bf bq[8];
#pragma unroll
  for (int kk = 0; kk < 8; ++kk)
    bq[kk] = *(const v16bf*)(f0base + (size_t)(wave * 16 + q) * C_DIM +
                             kk * 32 + half * 16);

  // This thread's 64B slice of each 8KB tile copy (4 x b128)
  const int cflat = tid * 64;             // 128 thr * 64B = 8KB
  const int crow  = cflat >> 9;           // /512
  const int coff  = cflat & 511;
  const unsigned ldsA = lds_off_u32(&tile[0][0]) + crow * ROW_PAD_B + coff;
  const unsigned ldsB = lds_off_u32(&tile[1][0]) + crow * ROW_PAD_B + coff;

  // Prologue: stage tile 0 into buffer 0
#pragma unroll
  for (int j = 0; j < 4; ++j)
    async_copy_b128(ldsA + j * 16, f1tile + cflat + j * 16);

  float m = -1e30f, Z = 0.f, Sx = 0.f, Sy = 0.f;

  for (int t = 0; t < STILES; ++t) {
    const unsigned ldsNext = ((t + 1) & 1) ? ldsB : ldsA;
    if (t + 1 < STILES) {
#pragma unroll
      for (int j = 0; j < 4; ++j)
        async_copy_b128(ldsNext + j * 16,
                        f1tile + (size_t)(t + 1) * TILE_BYTES + cflat + j * 16);
      asm volatile("s_wait_asynccnt 0x4" ::: "memory"); // tile t done, t+1 may fly
    } else {
      asm volatile("s_wait_asynccnt 0x0" ::: "memory");
    }
    __syncthreads();                       // tile t visible to all waves

    const __bf16* tb = &tile[t & 1][0];
    v8f acc = {};
#pragma unroll
    for (int kk = 0; kk < 8; ++kk) {
      const int k0 = kk * 32;
      union { v16bf v; v8bf h[2]; } af;
      af.h[0] = *(const v8bf*)(tb + q * ROW_PAD_E + k0 + half * 16);     // ds_load_b128
      af.h[1] = *(const v8bf*)(tb + q * ROW_PAD_E + k0 + half * 16 + 8); // ds_load_b128
      acc = __builtin_amdgcn_wmma_f32_16x16x32_bf16(
          false, af.v, false, bq[kk], (short)0, acc, false, false);
    }
    __syncthreads();                       // all consumed before t+2 overwrites

    // acc[v] = log2e * corr(s0+v, q),  s0 = t*16 + 8*half  (8-aligned)
    const int   s0   = t * 16 + 8 * half;
    const int   y0i  = s0 / HW;            // constant over the 8 scores
    const float remf = (float)(s0 - y0i * HW);
    const float y0f  = (float)y0i;

    const float tmax = fmaxf(fmaxf(fmaxf(acc[0], acc[1]), fmaxf(acc[2], acc[3])),
                             fmaxf(fmaxf(acc[4], acc[5]), fmaxf(acc[6], acc[7])));
    const float mn = fmaxf(m, tmax);
    const float r  = __builtin_exp2f(m - mn);
    float psum = 0.f, pv = 0.f;
#pragma unroll
    for (int v = 0; v < 8; ++v) {
      const float p = __builtin_exp2f(acc[v] - mn);
      psum += p;
      pv = fmaf(p, (float)v, pv);
    }
    m  = mn;
    Z  = fmaf(Z, r, psum);
    Sx = fmaf(Sx, r, fmaf(psum, remf, pv));   // Σ p*(rem+v)
    Sy = fmaf(Sy, r, y0f * psum);             // Σ p*y0
  }

  // Each query's two partials live in lanes q and q+16 of the same wave.
  const float m2  = __shfl_xor(m, 16);
  const float Z2  = __shfl_xor(Z, 16);
  const float Sx2 = __shfl_xor(Sx, 16);
  const float Sy2 = __shfl_xor(Sy, 16);
  const float M  = fmaxf(m, m2);
  const float rA = __builtin_exp2f(m - M);
  const float rB = __builtin_exp2f(m2 - M);
  const float Zt = Z * rA + Z2 * rB;
  const float Xt = Sx * rA + Sx2 * rB;
  const float Yt = Sy * rA + Sy2 * rB;

  if (lane < 16) {
    const float cx = Xt / Zt, cy = Yt / Zt;
    const int l  = qblk * 64 + wave * 16 + lane;
    const int y0 = l / HW, x0 = l % HW;
    float fx = cx - (float)x0;
    float fy = cy - (float)y0;
    if (y0 < 2 || y0 >= HW - 2 || x0 < 2 || x0 >= HW - 2) { fx = 0.f; fy = 0.f; }
    out[(((size_t)b * 2 + 0) * HW + y0) * HW + x0] = fx;
    out[(((size_t)b * 2 + 1) * HW + y0) * HW + x0] = fy;
  }
}

// ---------------------------------------------------------------------------
extern "C" void kernel_launch(void* const* d_in, const int* in_sizes, int n_in,
                              void* d_out, int out_size, void* d_ws, size_t ws_size,
                              hipStream_t stream) {
  const float* feat_c0 = (const float*)d_in[0];
  const float* feat_c1 = (const float*)d_in[1];
  const float* W       = (const float*)d_in[2];
  const float* bias    = (const float*)d_in[3];
  float* out = (float*)d_out;

  const size_t fElems = (size_t)NB * L_TOT * C_DIM;   // 4,718,592 bf16 elems
  __bf16* f0h = reinterpret_cast<__bf16*>(d_ws);
  __bf16* f1h = reinterpret_cast<__bf16*>((char*)d_ws + fElems * sizeof(__bf16));

  const float scale0 = INV_SQRT_C * INV_SQRT_C * LOG2E;  // dot(g0,f1)=log2e*corr
  const float scale1 = INV_SQRT_C;

  dim3 gA((NB * L_TOT) / 128, C_DIM / 16);   // 144 x 16
  proj_kernel<<<gA, 256, 0, stream>>>(feat_c0, W, bias, f0h, scale0);
  proj_kernel<<<gA, 256, 0, stream>>>(feat_c1, W, bias, f1h, scale1);

  dim3 gB(L_TOT / 64, NB);                   // 144 x 2, 128 thr (4 waves)
  corr_flow_kernel<<<gB, 128, 0, stream>>>(f0h, f1h, out);
}
